// Block_85375359910651
// MI455X (gfx1250) — matrix operand order repaired and checked by reading, not compiled, not run on hardware
//
#include <hip/hip_runtime.h>
#include <hip/hip_bf16.h>

typedef __attribute__((ext_vector_type(16))) _Float16 v16h;
typedef __attribute__((ext_vector_type(8)))  _Float16 v8h;
typedef __attribute__((ext_vector_type(4)))  _Float16 v4h;
typedef __attribute__((ext_vector_type(8)))  float    v8f;
typedef __attribute__((ext_vector_type(4)))  float    v4f;
typedef __attribute__((ext_vector_type(4)))  unsigned int u32x4;
typedef __attribute__((ext_vector_type(8)))  int      i32x8;
typedef __attribute__((ext_vector_type(4)))  int      i32x4;

#define BB 32
#define TT 500
#define CC 512
#define HH 8
#define DD 64

__device__ __forceinline__ v8f wmma_f16(v16h a, v16h b, v8f c) {
    // v_wmma_f32_16x16x32_f16: D = A(16x32 f16) * B(32x16 f16) + C(16x16 f32)
    return __builtin_amdgcn_wmma_f32_16x16x32_f16(false, a, false, b, (short)0, c, false, false);
}

// A-fragment from row-major f16 LDS tile (stride mult of 8 halfs).
// lane: m=lane&15, hs=lane>>4; k(e) = 8*hs+e (e<8), 16+8*hs+(e-8) (e>=8)
__device__ __forceinline__ v16h load_frag_a(const _Float16* p, int stride) {
    int lane = threadIdx.x & 31;
    int m = lane & 15, hs = lane >> 4;
    const _Float16* r0 = p + m * stride + 8 * hs;
    v8h lo = *(const v8h*)r0;
    v8h hi = *(const v8h*)(r0 + 16);
    v16h a;
#pragma unroll
    for (int i = 0; i < 8; ++i) { a[i] = lo[i]; a[i + 8] = hi[i]; }
    return a;
}

// B-fragment from K-transposed f16 LDS tile Bt[n][k] (stride mult of 8 halfs).
// lane: n=lane&15, hs=lane>>4; k(e) = 8*hs+e (e<8), 16+8*hs+(e-8) (e>=8)
__device__ __forceinline__ v16h load_frag_b(const _Float16* p, int stride) {
    int lane = threadIdx.x & 31;
    int n = lane & 15, hs = lane >> 4;
    const _Float16* r0 = p + n * stride + 8 * hs;
    v8h lo = *(const v8h*)r0;
    v8h hi = *(const v8h*)(r0 + 16);
    v16h b;
#pragma unroll
    for (int i = 0; i < 8; ++i) { b[i] = lo[i]; b[i + 8] = hi[i]; }
    return b;
}

// ---- Tensor Data Mover: 2-D tile load Global(f16) -> LDS, with LDS row padding ----
// D# packing per CDNA5 ISA ch.8 (groups 0/1; groups 2/3 zero for 2-D tiles).
// 6-arg builtin form (clang-23 / therock headers): (g0, g1, g2, g3, g4, cpol)
__device__ __forceinline__ void tdm_load_2d(unsigned lds_addr, const void* gptr,
                                            unsigned tile_d0, unsigned tile_d1,
                                            unsigned tdim0, unsigned tdim1,
                                            unsigned stride0,
                                            unsigned pad_interval_code,
                                            unsigned pad_amount_code) {
    unsigned long long ga = (unsigned long long)(uintptr_t)gptr;
    u32x4 g0;
    g0[0] = 1u;                                       // count=1, user descriptor
    g0[1] = lds_addr;                                 // LDS byte address
    g0[2] = (unsigned)(ga & 0xffffffffu);             // global_addr[31:0]
    g0[3] = (unsigned)((ga >> 32) & 0x01ffffffu) | 0x80000000u;  // addr[56:32], type=2
    i32x8 g1;
    g1[0] = (int)((1u << 16) | (1u << 20) | (pad_interval_code << 22) |
                  (pad_amount_code << 25));           // data_size=2B, pad_enable
    g1[1] = (int)((tdim0 & 0xffffu) << 16);           // tensor_dim0[15:0]
    g1[2] = (int)(((tdim0 >> 16) & 0xffffu) | ((tdim1 & 0xffffu) << 16));
    g1[3] = (int)(((tdim1 >> 16) & 0xffffu) | ((tile_d0 & 0xffffu) << 16));
    g1[4] = (int)(tile_d1 & 0xffffu);                 // tile_dim1, tile_dim2=0
    g1[5] = (int)stride0;                             // tensor_dim0_stride[31:0]
    g1[6] = 0;
    g1[7] = 0;
    i32x4 z4 = {0, 0, 0, 0};
    i32x8 z8 = {0, 0, 0, 0, 0, 0, 0, 0};
    __builtin_amdgcn_tensor_load_to_lds(g0, g1, z4, z4, z8, 0);
}

// ---------------- bucket ids (B,T,T) u8 ----------------
__global__ __launch_bounds__(256) void rpe_bucket_kernel(const int* __restrict__ ct,
                                                         unsigned char* __restrict__ bucket) {
    long long i = (long long)blockIdx.x * 256 + threadIdx.x;
    if (i >= (long long)BB * TT * TT) return;
    int s = (int)(i % TT);
    int t = (int)((i / TT) % TT);
    int b = (int)(i / ((long long)TT * TT));
    const int* cb = ct + (size_t)b * TT * 2;
    float dx = (float)(cb[t * 2 + 0] - cb[s * 2 + 0]);
    float dy = (float)(cb[t * 2 + 1] - cb[s * 2 + 1]);
    float dist = sqrtf(dx * dx + dy * dy);
    float rel = truncf(dist / 12.0f);
    float abs_r = fabsf(rel);
    float lb = copysignf(
        fminf(rintf(12.0f + logf(fmaxf(abs_r, 1e-6f) / 12.0f) * (12.0f / 2.0794415417f)), 24.0f),
        rel);
    float idx = (abs_r <= 12.0f) ? rintf(rel) : lb;
    bucket[i] = (unsigned char)((int)idx + 24);
}

// ---------------- generic GEMM: C = act(A*B + bias), f16 WMMA, f32 accum ---------
// A: MxK f32 row-major, B: KxN f32 row-major. BM=128, BN=128, BK=32, 256 thr (8 waves)
template <bool GELU, typename OutT>
__global__ __launch_bounds__(256) void gemm_kernel(const float* __restrict__ A,
                                                   const float* __restrict__ Bm,
                                                   const float* __restrict__ bias,
                                                   OutT* __restrict__ C,
                                                   int M, int N, int K) {
    __shared__ _Float16 sA[128 * 40];
    __shared__ _Float16 sBt[128 * 40];

    int tid = threadIdx.x;
    int wid = tid >> 5;
    int lane = tid & 31;
    int m0 = blockIdx.y * 128;
    int n0 = blockIdx.x * 128;
    int wm = wid >> 1;
    int wn = wid & 1;

    v8f acc[2][4];
#pragma unroll
    for (int i = 0; i < 2; ++i)
#pragma unroll
        for (int j = 0; j < 4; ++j) acc[i][j] = (v8f)0.0f;

    int arow = tid >> 1, aseg = (tid & 1) * 16;
    int bn = tid & 127, bkh = tid >> 7;

    for (int k0 = 0; k0 < K; k0 += 32) {
        __syncthreads();
        {   // stage A tile 128x32 -> f16
            const float* gp = A + (size_t)(m0 + arow) * K + k0 + aseg;
#pragma unroll
            for (int i = 0; i < 4; ++i) {
                v4f f = *(const v4f*)(gp + 4 * i);
                v4h hh;
#pragma unroll
                for (int q = 0; q < 4; ++q) hh[q] = (_Float16)f[q];
                *(v4h*)(sA + arow * 40 + aseg + 4 * i) = hh;
            }
        }
        {   // stage B tile 32x128 transposed -> sBt[n][k]
            const float* gp = Bm + (size_t)(k0 + bkh * 16) * N + n0 + bn;
            v8h h0, h1;
#pragma unroll
            for (int kk = 0; kk < 8; ++kk) h0[kk] = (_Float16)gp[(size_t)kk * N];
#pragma unroll
            for (int kk = 0; kk < 8; ++kk) h1[kk] = (_Float16)gp[(size_t)(kk + 8) * N];
            *(v8h*)(sBt + bn * 40 + bkh * 16) = h0;
            *(v8h*)(sBt + bn * 40 + bkh * 16 + 8) = h1;
        }
        __syncthreads();

        v16h af[2], bf[4];
#pragma unroll
        for (int i = 0; i < 2; ++i) af[i] = load_frag_a(sA + (wm * 32 + i * 16) * 40, 40);
#pragma unroll
        for (int j = 0; j < 4; ++j) bf[j] = load_frag_b(sBt + (wn * 64 + j * 16) * 40, 40);
#pragma unroll
        for (int i = 0; i < 2; ++i)
#pragma unroll
            for (int j = 0; j < 4; ++j) acc[i][j] = wmma_f16(af[i], bf[j], acc[i][j]);
    }

    int cbase = n0 + wn * 64 + (lane & 15);
    int rbase = m0 + wm * 32 + ((lane >> 4) << 3);
#pragma unroll
    for (int i = 0; i < 2; ++i) {
#pragma unroll
        for (int j = 0; j < 4; ++j) {
            int col = cbase + j * 16;
            float bv = bias[col];
#pragma unroll
            for (int e = 0; e < 8; ++e) {
                int row = rbase + i * 16 + e;
                float v = acc[i][j][e] + bv;
                if (GELU) v = 0.5f * v * (1.0f + erff(v * 0.70710678118f));
                C[(size_t)row * N + col] = (OutT)v;
            }
        }
    }
}

// ------------- V transpose: qkv16 (B,T,1536) f16 -> v16t (B,H,D,T) f16 -------------
// grid (16 t-tiles, H, B), block 256; tile 32t x 64d via LDS
__global__ __launch_bounds__(256) void vtrans_kernel(const _Float16* __restrict__ qkv16,
                                                     _Float16* __restrict__ v16t) {
    __shared__ _Float16 tile[32 * 72];
    int tid = threadIdx.x;
    int b = blockIdx.z, h = blockIdx.y;
    int t0 = blockIdx.x * 32;
    {
        int tl = tid >> 3, dseg = (tid & 7) * 8;
        int t = t0 + tl;
        v8h v;
        if (t < TT) {
            v = *(const v8h*)(qkv16 + ((size_t)b * TT + t) * (3 * CC) + 2 * CC + h * DD + dseg);
        } else {
            #pragma unroll
            for (int q = 0; q < 8; ++q) v[q] = (_Float16)0.0f;
        }
        *(v8h*)(tile + tl * 72 + dseg) = v;
    }
    __syncthreads();
    {
        int d = tid >> 2, tseg = (tid & 2 ? 16 : 0) + (tid & 1) * 8;
        _Float16* op = v16t + ((size_t)(b * HH + h) * DD + d) * TT + t0 + tseg;
#pragma unroll
        for (int i = 0; i < 8; ++i) {
            int t = t0 + tseg + i;
            if (t < TT) op[i] = tile[(tseg + i) * 72 + d];
        }
    }
}

// ---------------- flash attention with RPE bias, TDM-staged K/V ----------------
// grid: (8 qblocks of 64, H, B); block 128 = 4 waves; wave w: queries q0+16w..+15
__global__ __launch_bounds__(128) void attn_kernel(const _Float16* __restrict__ qkv16,
                                                   const _Float16* __restrict__ v16t,
                                                   const unsigned char* __restrict__ bucket,
                                                   const float* __restrict__ rpe_table,
                                                   float* __restrict__ y) {
    __shared__ _Float16 sK[2][32 * 72];     // K tile [key][d], stride 72 (TDM pad)
    __shared__ _Float16 sV[2][64 * 40];     // V tile transposed [d][key], stride 40 (TDM pad)
    __shared__ _Float16 sP[4 * 16 * 40];    // per-wave P tile [q][key], stride 40
    __shared__ float sTab[52];

    int tid = threadIdx.x;
    int wid = tid >> 5, lane = tid & 31;
    int b = blockIdx.z, h = blockIdx.y;
    int q0 = blockIdx.x * 64;
    int hh = (8 * b + h) >> 5;   // faithful torch reshape: table row
    int bb = (8 * b + h) & 31;   // faithful torch reshape: coord batch

    if (tid < 49) sTab[tid] = rpe_table[hh * 49 + tid];

    int m = lane & 15, hs = lane >> 4;
    int tq0 = q0 + wid * 16;

    const _Float16* kbase = qkv16 + (size_t)b * TT * (3 * CC) + CC + h * DD;
    const _Float16* vbase = v16t + (size_t)(b * HH + h) * DD * TT;

    // prefetch chunk 0 via TDM (wave 0 issues; EXEC-independent DMA)
    if (wid == 0) {
        tdm_load_2d((unsigned)(uintptr_t)&sK[0][0], kbase, 64, 32, 64, (unsigned)TT,
                    3 * CC, 4, 3);
        tdm_load_2d((unsigned)(uintptr_t)&sV[0][0], vbase, 32, 64, (unsigned)TT, 64,
                    (unsigned)TT, 3, 3);
    }

    // Q fragments for D=64 (2 k-chunks of 32) straight from f16 qkv
    v16h qa[2];
    {
        int tq = tq0 + m;
        int tcq = tq < TT ? tq : TT - 1;
        const _Float16* qp = qkv16 + ((size_t)b * TT + tcq) * (3 * CC) + h * DD;
#pragma unroll
        for (int c = 0; c < 2; ++c) {
            const _Float16* r0 = qp + c * 32 + 8 * hs;
            v8h lo = *(const v8h*)r0;
            v8h hi = *(const v8h*)(r0 + 16);
            v16h a;
#pragma unroll
            for (int q = 0; q < 8; ++q) { a[q] = lo[q]; a[q + 8] = hi[q]; }
            qa[c] = a;
        }
    }

    v8f o[4];
#pragma unroll
    for (int d = 0; d < 4; ++d) o[d] = (v8f)0.0f;
    float mrun[8], lrun[8];
#pragma unroll
    for (int e = 0; e < 8; ++e) { mrun[e] = -3.0e38f; lrun[e] = 0.0f; }

    _Float16* pw = sP + wid * 640;

    for (int kc = 0; kc < 16; ++kc) {
        int cur = kc & 1;
        int s0 = kc * 32;
        // issue next chunk into the other buffer while we compute this one
        if (wid == 0 && kc < 15) {
            int sn = s0 + 32;
            unsigned rem = (unsigned)(TT - sn);
            tdm_load_2d((unsigned)(uintptr_t)&sK[cur ^ 1][0], kbase + (size_t)sn * (3 * CC),
                        64, 32, 64, rem, 3 * CC, 4, 3);
            tdm_load_2d((unsigned)(uintptr_t)&sV[cur ^ 1][0], vbase + sn,
                        32, 64, rem, 64, (unsigned)TT, 3, 3);
        }
        if (kc < 15) __builtin_amdgcn_s_wait_tensorcnt(2);
        else         __builtin_amdgcn_s_wait_tensorcnt(0);
        __syncthreads();

        // S = Q K^T for 32 keys (two 16x16 tiles), matmul K-dim = D = 64
        v8f sa = (v8f)0.0f, sb = (v8f)0.0f;
#pragma unroll
        for (int c = 0; c < 2; ++c) {
            v16h b0 = load_frag_b(&sK[cur][0] + c * 32, 72);
            v16h b1 = load_frag_b(&sK[cur][0] + 16 * 72 + c * 32, 72);
            sa = wmma_f16(qa[c], b0, sa);
            sb = wmma_f16(qa[c], b1, sb);
        }

        int col0 = s0 + (lane & 15);
        int col1 = col0 + 16;
#pragma unroll
        for (int e = 0; e < 8; ++e) {
            int r = e + (hs << 3);
            int t = tq0 + r; int tc = t < TT ? t : TT - 1;
            size_t brow = (size_t)bb * TT * TT + (size_t)tc * TT;
            float va = (col0 < TT) ? (sa[e] * 0.125f + sTab[bucket[brow + col0]]) : -1.0e30f;
            float vb = (col1 < TT) ? (sb[e] * 0.125f + sTab[bucket[brow + col1]]) : -1.0e30f;

            float mx = fmaxf(va, vb);
#pragma unroll
            for (int off = 1; off < 16; off <<= 1) mx = fmaxf(mx, __shfl_xor(mx, off, 16));
            float mnew = fmaxf(mrun[e], mx);
            float sc = __expf(mrun[e] - mnew);
            float pa = __expf(va - mnew);
            float pb = __expf(vb - mnew);
            float rs = pa + pb;
#pragma unroll
            for (int off = 1; off < 16; off <<= 1) rs += __shfl_xor(rs, off, 16);
            lrun[e] = lrun[e] * sc + rs;
            mrun[e] = mnew;
#pragma unroll
            for (int d = 0; d < 4; ++d) o[d][e] *= sc;
            pw[r * 40 + (lane & 15)] = (_Float16)pa;
            pw[r * 40 + 16 + (lane & 15)] = (_Float16)pb;
        }
        __syncthreads();

        // O += P V (matmul K-dim = 32 keys)
        v16h pf = load_frag_a(pw, 40);
#pragma unroll
        for (int d = 0; d < 4; ++d) {
            v16h vf = load_frag_b(&sV[cur][0] + d * 16 * 40, 40);
            o[d] = wmma_f16(pf, vf, o[d]);
        }
        __syncthreads();   // reads of buf[cur] done before it is re-issued next iter
    }

    // normalize and store (B,T,H*D) f32
#pragma unroll
    for (int e = 0; e < 8; ++e) {
        float inv = 1.0f / lrun[e];
        int r = e + (hs << 3);
        int t = tq0 + r;
        if (t < TT) {
            size_t base = ((size_t)b * TT + t) * CC + h * DD + (lane & 15);
#pragma unroll
            for (int d = 0; d < 4; ++d) y[base + d * 16] = o[d][e] * inv;
        }
    }
}

// ---------------- residual + LayerNorm (one row of C=512 per block) ----------------
__global__ __launch_bounds__(256) void resln_kernel(const float* __restrict__ a,
                                                    const float* __restrict__ r,
                                                    const float* __restrict__ w,
                                                    const float* __restrict__ bvec,
                                                    float* __restrict__ out) {
    __shared__ float sred[256];
    int row = blockIdx.x, tid = threadIdx.x;
    const float* pa = a + (size_t)row * CC;
    const float* pr = r + (size_t)row * CC;
    float v0 = pa[tid] + pr[tid];
    float v1 = pa[tid + 256] + pr[tid + 256];
    sred[tid] = v0 + v1;
    __syncthreads();
    for (int o = 128; o > 0; o >>= 1) {
        if (tid < o) sred[tid] += sred[tid + o];
        __syncthreads();
    }
    float mean = sred[0] * (1.0f / 512.0f);
    __syncthreads();
    float d0 = v0 - mean, d1 = v1 - mean;
    sred[tid] = d0 * d0 + d1 * d1;
    __syncthreads();
    for (int o = 128; o > 0; o >>= 1) {
        if (tid < o) sred[tid] += sred[tid + o];
        __syncthreads();
    }
    float rstd = rsqrtf(sred[0] * (1.0f / 512.0f) + 1e-5f);
    out[(size_t)row * CC + tid] = d0 * rstd * w[tid] + bvec[tid];
    out[(size_t)row * CC + tid + 256] = d1 * rstd * w[tid + 256] + bvec[tid + 256];
}

extern "C" void kernel_launch(void* const* d_in, const int* in_sizes, int n_in,
                              void* d_out, int out_size, void* d_ws, size_t ws_size,
                              hipStream_t stream) {
    const float* x       = (const float*)d_in[0];
    const int*   ct      = (const int*)d_in[1];
    const float* w_attn  = (const float*)d_in[2];
    const float* b_attn  = (const float*)d_in[3];
    const float* w_proj  = (const float*)d_in[4];
    const float* b_proj  = (const float*)d_in[5];
    const float* rpe_tab = (const float*)d_in[6];
    const float* ln1_w   = (const float*)d_in[7];
    const float* ln1_b   = (const float*)d_in[8];
    const float* ln2_w   = (const float*)d_in[9];
    const float* ln2_b   = (const float*)d_in[10];
    const float* w_fc    = (const float*)d_in[11];
    const float* b_fc    = (const float*)d_in[12];
    const float* w_fc2   = (const float*)d_in[13];
    const float* b_fc2   = (const float*)d_in[14];

    const int M = BB * TT;  // 16000
    char* ws = (char*)d_ws;
    size_t off = 0;
    _Float16* qkv16 = (_Float16*)(ws + off); off += (size_t)M * 3 * CC * 2;        // 49.2 MB
    _Float16* v16t  = (_Float16*)(ws + off); off += (size_t)BB * HH * DD * TT * 2; // 16.4 MB
    float* ybuf = (float*)(ws + off); off += (size_t)M * CC * 4;                   // 32.8 MB
    float* pbuf = (float*)(ws + off); off += (size_t)M * CC * 4;                   // 32.8 MB
    float* xln  = (float*)(ws + off); off += (size_t)M * CC * 4;                   // 32.8 MB
    float* hbuf = (float*)(ws + off); off += (size_t)M * 2 * CC * 4;               // 65.5 MB
    unsigned char* bucket = (unsigned char*)(ws + off);                            // 8 MB

    // 1. RPE bucket ids
    {
        long long n = (long long)BB * TT * TT;
        rpe_bucket_kernel<<<dim3((unsigned)((n + 255) / 256)), 256, 0, stream>>>(ct, bucket);
    }
    // 2. QKV GEMM (16000x512)*(512x1536), f16 output for WMMA/TDM consumption
    gemm_kernel<false, _Float16><<<dim3(3 * CC / 128, M / 128), 256, 0, stream>>>(
        x, w_attn, b_attn, qkv16, M, 3 * CC, CC);
    // 3. V transpose to (B,H,D,T) so TDM tiles are inner-contiguous
    vtrans_kernel<<<dim3(16, HH, BB), 256, 0, stream>>>(qkv16, v16t);
    // 4. flash attention with RPE bias, TDM double-buffered K/V staging
    attn_kernel<<<dim3(8, HH, BB), 128, 0, stream>>>(qkv16, v16t, bucket, rpe_tab, ybuf);
    // 5. output projection
    gemm_kernel<false, float><<<dim3(CC / 128, M / 128), 256, 0, stream>>>(
        ybuf, w_proj, b_proj, pbuf, M, CC, CC);
    // 6. residual + LN1 -> xln
    resln_kernel<<<M, 256, 0, stream>>>(x, pbuf, ln1_w, ln1_b, xln);
    // 7. FC with exact GELU
    gemm_kernel<true, float><<<dim3(2 * CC / 128, M / 128), 256, 0, stream>>>(
        xln, w_fc, b_fc, hbuf, M, 2 * CC, CC);
    // 8. FC2
    gemm_kernel<false, float><<<dim3(CC / 128, M / 128), 256, 0, stream>>>(
        hbuf, w_fc2, b_fc2, pbuf, M, CC, 2 * CC);
    // 9. residual + LN2 -> out
    resln_kernel<<<M, 256, 0, stream>>>(xln, pbuf, ln2_w, ln2_b, (float*)d_out);
}
